// AttentionHead_21337397526972
// MI455X (gfx1250) — matrix operand order repaired and checked by reading, not compiled, not run on hardware
//
#include <hip/hip_runtime.h>

typedef __bf16 bf16;
typedef __attribute__((ext_vector_type(16))) __bf16 v16bf;
typedef __attribute__((ext_vector_type(8)))  __bf16 bf16x8;
typedef __attribute__((ext_vector_type(8)))  float  v8f;

#define B_ 8
#define T_ 2048
#define C_ 1024
#define H_ 128

#if __has_builtin(__builtin_amdgcn_global_load_async_to_lds_b128) && \
    __has_builtin(__builtin_amdgcn_s_wait_asynccnt)
#define USE_ASYNC_LDS 1
#else
#define USE_ASYNC_LDS 0
#endif

// Async-LDS builtin takes pointers to 16-byte int vectors (per hipcc diag).
typedef int v4i_vs __attribute__((vector_size(16)));
typedef __attribute__((address_space(1))) v4i_vs g_v4i;   // global src
typedef __attribute__((address_space(3))) v4i_vs l_v4i;   // LDS dst

// ---------------------------------------------------------------------------
// Kernel 0: pre-transpose W{q,k,v} fp32 [C][H] -> bf16 Wt [3][H][C].
// Makes the GEMM B-operand a contiguous 32-byte load per lane.
// 3*H*C/8 = 49152 chunks of 8, 192 blocks x 256 threads.
// ---------------------------------------------------------------------------
__global__ __launch_bounds__(256) void prep_w_kernel(
    const float* __restrict__ Wq, const float* __restrict__ Wk,
    const float* __restrict__ Wv, bf16* __restrict__ Wt)
{
    const int id  = blockIdx.x * 256 + threadIdx.x;   // 0..49151
    const int mat = id / (H_ * C_ / 8);
    const int rem = id % (H_ * C_ / 8);
    const int n   = rem / (C_ / 8);
    const int kc  = (rem % (C_ / 8)) * 8;

    const float* W = (mat == 0) ? Wq : (mat == 1) ? Wk : Wv;
    bf16* dst = Wt + ((size_t)mat * H_ + n) * C_ + kc;
    #pragma unroll
    for (int j = 0; j < 8; ++j)
        dst[j] = (bf16)W[(size_t)(kc + j) * H_ + n];  // transpose: [k][n] -> [n][k]
}

// ---------------------------------------------------------------------------
// Kernel 1: q/k/v = x @ W + b as bf16 (fp32 accumulate).
// One wave computes a 16x128 strip (8 n-tiles) of one matrix: each A-operand
// load feeds 8 WMMAs. 1024 row-tiles x 3 mats = 3072 waves = 384 blocks.
// ---------------------------------------------------------------------------
__global__ __launch_bounds__(256) void qkv_proj_kernel(
    const float* __restrict__ x,  const bf16* __restrict__ Wt,
    const float* __restrict__ bq, const float* __restrict__ bk,
    const float* __restrict__ bv,
    bf16* __restrict__ Qw, bf16* __restrict__ Kw, bf16* __restrict__ Vw)
{
    const int lane  = threadIdx.x & 31;
    const int wave  = threadIdx.x >> 5;
    const int gwave = blockIdx.x * 8 + wave;

    const int rowTile = gwave / 3;           // 0..1023
    const int mat     = gwave % 3;           // 0:q 1:k 2:v
    const int row0    = rowTile * 16;

    const float* bias = (mat == 0) ? bq : (mat == 1) ? bk : bv;
    bf16* dst         = (mat == 0) ? Qw : (mat == 1) ? Kw : Vw;
    const bf16* Wm    = Wt + (size_t)mat * H_ * C_;   // [H][C] bf16

    const int n    = lane & 15;              // A row / B col / D col
    const int hi   = lane >> 4;              // lane half
    const int aoff = hi * 16;                // A-layout K offset
    const int koff = hi * 16;                // B-layout K offset

    v8f acc[8];
    #pragma unroll
    for (int i = 0; i < 8; ++i) acc[i] = {};

    for (int k0 = 0; k0 < C_; k0 += 32) {
        // A operand: row (row0+n), K slots {aoff..aoff+15}
        const float* ap = x + (size_t)(row0 + n) * C_ + k0 + aoff;
        v16bf a;
        #pragma unroll
        for (int j = 0; j < 8; ++j) {
            a[j]     = (bf16)ap[j];
            a[j + 8] = (bf16)ap[8 + j];
        }
        // 8 n-tiles: B operand contiguous from transposed W
        #pragma unroll
        for (int nt = 0; nt < 8; ++nt) {
            const bf16* bp = Wm + (size_t)(nt * 16 + n) * C_ + k0 + koff;
            v16bf b = *(const v16bf*)bp;
            acc[nt] = __builtin_amdgcn_wmma_f32_16x16x32_bf16(
                          false, a, false, b, (short)0, acc[nt], false, false);
        }
    }

    #pragma unroll
    for (int nt = 0; nt < 8; ++nt) {
        const float bb = bias[nt * 16 + n];
        #pragma unroll
        for (int r = 0; r < 8; ++r) {
            const int row = row0 + r + 8 * hi;     // D layout
            dst[(size_t)row * H_ + nt * 16 + n] = (bf16)(acc[nt][r] + bb);
        }
    }
}

// ---------------------------------------------------------------------------
// Kernel 2: flash attention. Block = 8 waves = 128 query rows of one batch.
// K tile staged via async global->LDS (ASYNCcnt) when available; V tile
// transposed through registers. Per-wave 16x128 fp32 accumulator + (m,l).
// ---------------------------------------------------------------------------
__global__ __launch_bounds__(256) void flash_attn_kernel(
    const bf16* __restrict__ Qw, const bf16* __restrict__ Kw,
    const bf16* __restrict__ Vw, float* __restrict__ out)
{
    __shared__ __align__(32) bf16 Kl[32][128];     // key tile, row-major
    __shared__ __align__(32) bf16 Vt[128][32];     // value tile, transposed
    __shared__ __align__(32) bf16 Pl[8][16][32];   // per-wave P staging

    const int tid  = threadIdx.x;
    const int lane = tid & 31;
    const int wave = tid >> 5;
    const int bat  = blockIdx.x >> 4;              // 16 blocks per batch
    const int tblk = blockIdx.x & 15;
    const size_t bbase = (size_t)bat * T_;         // batch row base in [B*T)
    const int q0t  = tblk * 128 + wave * 16;       // first query t of this wave
    const int n    = lane & 15;
    const int hi   = lane >> 4;
    const float scale = 0.08838834764831845f;      // 1/sqrt(128)

    // Preload Q tile (16 rows x H) as 4 A-operands.
    v16bf qreg[4];
    {
        const int aoff = hi * 16;
        #pragma unroll
        for (int hc = 0; hc < 4; ++hc) {
            const bf16* qp = Qw + (bbase + q0t + n) * H_ + hc * 32 + aoff;
            bf16x8 lo = *(const bf16x8*)qp;
            bf16x8 hh = *(const bf16x8*)(qp + 8);
            #pragma unroll
            for (int j = 0; j < 8; ++j) { qreg[hc][j] = lo[j]; qreg[hc][j + 8] = hh[j]; }
        }
    }

    v8f acc[8];
    #pragma unroll
    for (int i = 0; i < 8; ++i) acc[i] = {};
    float mrow[8], lrow[8];
    #pragma unroll
    for (int r = 0; r < 8; ++r) { mrow[r] = -__builtin_inff(); lrow[r] = 0.f; }

    const int kend = tblk * 128 + 128;             // exclusive key bound
    for (int k0 = 0; k0 < kend; k0 += 32) {
        // ---- stage K tile: async global->LDS (b128 = 8 bf16 per chunk) ----
#if USE_ASYNC_LDS
        #pragma unroll
        for (int i = 0; i < 2; ++i) {
            const int c = i * 256 + tid;           // 512 chunks of 8 elems
            const int r = c >> 4, h = (c & 15) * 8;
            __builtin_amdgcn_global_load_async_to_lds_b128(
                (g_v4i*)(Kw + (bbase + k0 + r) * H_ + h),
                (l_v4i*)&Kl[r][h], 0, 0);
        }
#else
        #pragma unroll
        for (int i = 0; i < 2; ++i) {
            const int c = i * 256 + tid;
            const int r = c >> 4, h = (c & 15) * 8;
            *(bf16x8*)&Kl[r][h] = *(const bf16x8*)(Kw + (bbase + k0 + r) * H_ + h);
        }
#endif
        // ---- stage V tile transposed (vector load, strided LDS store) ----
        #pragma unroll
        for (int i = 0; i < 2; ++i) {
            const int c = i * 256 + tid;
            const int r = c >> 4, h0 = (c & 15) * 8;
            bf16x8 vv = *(const bf16x8*)(Vw + (bbase + k0 + r) * H_ + h0);
            #pragma unroll
            for (int j = 0; j < 8; ++j) Vt[h0 + j][r] = vv[j];
        }
#if USE_ASYNC_LDS
        __builtin_amdgcn_s_wait_asynccnt(0);
#endif
        __syncthreads();

        if (k0 <= q0t + 15) {
            // ---- S = Q K^T (two 16x16 tiles covering 32 keys) ----
            v8f s[2];
            #pragma unroll
            for (int nt = 0; nt < 2; ++nt) {
                v8f sa = {};
                #pragma unroll
                for (int hc = 0; hc < 4; ++hc) {
                    const bf16* bp = &Kl[nt * 16 + n][hc * 32 + hi * 16];
                    v16bf bm = *(const v16bf*)bp;
                    sa = __builtin_amdgcn_wmma_f32_16x16x32_bf16(
                             false, qreg[hc], false, bm, (short)0, sa, false, false);
                }
                s[nt] = sa;
            }

            // ---- scale + causal mask + row max ----
            float rmax[8];
            #pragma unroll
            for (int r = 0; r < 8; ++r) {
                const int qt = q0t + r + 8 * hi;
                float v0 = s[0][r] * scale;
                float v1 = s[1][r] * scale;
                if (k0 + n      > qt) v0 = -__builtin_inff();
                if (k0 + 16 + n > qt) v1 = -__builtin_inff();
                s[0][r] = v0; s[1][r] = v1;
                float mx = fmaxf(v0, v1);
                #pragma unroll
                for (int off = 1; off < 16; off <<= 1)   // stays in 16-lane half
                    mx = fmaxf(mx, __shfl_xor(mx, off, 32));
                rmax[r] = mx;
            }

            float alpha[8];
            #pragma unroll
            for (int r = 0; r < 8; ++r) {
                const float mnew = fmaxf(mrow[r], rmax[r]);
                alpha[r] = (mnew == -__builtin_inff()) ? 1.0f : __expf(mrow[r] - mnew);
                mrow[r] = mnew;
            }

            // ---- P = exp(S - m), row sums, stage P as bf16 in LDS ----
            #pragma unroll
            for (int r = 0; r < 8; ++r) {
                const int qt = q0t + r + 8 * hi;
                const float e0 = (k0 + n      <= qt) ? __expf(s[0][r] - mrow[r]) : 0.f;
                const float e1 = (k0 + 16 + n <= qt) ? __expf(s[1][r] - mrow[r]) : 0.f;
                Pl[wave][r + 8 * hi][n]      = (bf16)e0;
                Pl[wave][r + 8 * hi][16 + n] = (bf16)e1;
                float sum = e0 + e1;
                #pragma unroll
                for (int off = 1; off < 16; off <<= 1)
                    sum += __shfl_xor(sum, off, 32);
                lrow[r] = lrow[r] * alpha[r] + sum;
            }

            // ---- rescale accumulator ----
            #pragma unroll
            for (int ht = 0; ht < 8; ++ht)
                #pragma unroll
                for (int r = 0; r < 8; ++r) acc[ht][r] *= alpha[r];

            // ---- reload P in A-layout (same-wave LDS ops are in-order) ----
            v16bf pa;
            {
                const int aoff = hi * 16;
                const bf16* pp = &Pl[wave][n][aoff];
                bf16x8 lo = *(const bf16x8*)pp;
                bf16x8 hh = *(const bf16x8*)(pp + 8);
                #pragma unroll
                for (int j = 0; j < 8; ++j) { pa[j] = lo[j]; pa[j + 8] = hh[j]; }
            }

            // ---- acc += P @ V (8 h-tiles, K-depth 32 = one WMMA each) ----
            #pragma unroll
            for (int ht = 0; ht < 8; ++ht) {
                const bf16* vp = &Vt[ht * 16 + n][hi * 16];
                v16bf vb = *(const v16bf*)vp;
                acc[ht] = __builtin_amdgcn_wmma_f32_16x16x32_bf16(
                              false, pa, false, vb, (short)0, acc[ht], false, false);
            }
        }
        __syncthreads();
    }

    // ---- epilogue: out = acc / l ----
    #pragma unroll
    for (int ht = 0; ht < 8; ++ht) {
        #pragma unroll
        for (int r = 0; r < 8; ++r) {
            const int row = q0t + r + 8 * hi;
            out[(bbase + row) * H_ + ht * 16 + n] = acc[ht][r] / lrow[r];
        }
    }
}

// ---------------------------------------------------------------------------
extern "C" void kernel_launch(void* const* d_in, const int* in_sizes, int n_in,
                              void* d_out, int out_size, void* d_ws, size_t ws_size,
                              hipStream_t stream) {
    const float* x  = (const float*)d_in[0];
    const float* Wq = (const float*)d_in[1];
    const float* bq = (const float*)d_in[2];
    const float* Wk = (const float*)d_in[3];
    const float* bk = (const float*)d_in[4];
    const float* Wv = (const float*)d_in[5];
    const float* bv = (const float*)d_in[6];
    float* out = (float*)d_out;

    bf16* Qw = (bf16*)d_ws;                           // 4 MB
    bf16* Kw = Qw + (size_t)B_ * T_ * H_;             // 4 MB
    bf16* Vw = Kw + (size_t)B_ * T_ * H_;             // 4 MB
    bf16* Wt = Vw + (size_t)B_ * T_ * H_;             // 768 KB: [3][H][C] bf16

    prep_w_kernel<<<192, 256, 0, stream>>>(Wq, Wk, Wv, Wt);
    // 1024 row tiles * 3 mats, 8 waves/block
    qkv_proj_kernel<<<384, 256, 0, stream>>>(x, Wt, bq, bk, bv, Qw, Kw, Vw);
    // 8 batches * 16 blocks of 128 query rows
    flash_attn_kernel<<<128, 256, 0, stream>>>(Qw, Kw, Vw, out);
}
